// Res18_Net_36816459661967
// MI455X (gfx1250) — 8-bit matrix operand order repaired and checked by reading, compiled, not run on hardware
//
#include <hip/hip_runtime.h>
#include <cstdint>

typedef __attribute__((ext_vector_type(8))) int v8i;
typedef __attribute__((ext_vector_type(2))) int v2i;

#define TPB 256
#define BN_EPS 1e-5f

__global__ __launch_bounds__(TPB) void k_stats_nchw(
    const float* __restrict__ x, int N, int C, int HW, float* __restrict__ stats)
{
    __shared__ float ss[TPB], sq[TPB];
    int c = blockIdx.x;
    int Mtot = N * HW;
    float s = 0.f, q = 0.f;
    for (int t = threadIdx.x; t < Mtot; t += TPB) {
        int n = t / HW, j = t - n * HW;
        float v = x[((long)n * C + c) * HW + j];
        s += v; q += v * v;
    }
    ss[threadIdx.x] = s; sq[threadIdx.x] = q;
    __syncthreads();
    for (int off = TPB / 2; off > 0; off >>= 1) {
        if (threadIdx.x < off) {
            ss[threadIdx.x] += ss[threadIdx.x + off];
            sq[threadIdx.x] += sq[threadIdx.x + off];
        }
        __syncthreads();
    }
    if (threadIdx.x == 0) {
        float mean = ss[0] / (float)Mtot;
        stats[c]     = mean;
        stats[C + c] = sq[0] / (float)Mtot - mean * mean;
    }
}

__global__ __launch_bounds__(TPB) void k_stats(
    const float* __restrict__ buf, int Mtot, int C, float* __restrict__ stats)
{
    __shared__ float ss[TPB], sq[TPB];
    int c = blockIdx.x;
    float s = 0.f, q = 0.f;
    for (int m = threadIdx.x; m < Mtot; m += TPB) {
        float v = buf[(long)m * C + c];
        s += v; q += v * v;
    }
    ss[threadIdx.x] = s; sq[threadIdx.x] = q;
    __syncthreads();
    for (int off = TPB / 2; off > 0; off >>= 1) {
        if (threadIdx.x < off) {
            ss[threadIdx.x] += ss[threadIdx.x + off];
            sq[threadIdx.x] += sq[threadIdx.x + off];
        }
        __syncthreads();
    }
    if (threadIdx.x == 0) {
        float mean = ss[0] / (float)Mtot;
        stats[c]     = mean;
        stats[C + c] = sq[0] / (float)Mtot - mean * mean;
    }
}

__global__ __launch_bounds__(TPB) void k_conv1(
    const float* __restrict__ x, const float* __restrict__ w,
    const float* __restrict__ xstats, float* __restrict__ out)
{
    long i = (long)blockIdx.x * TPB + threadIdx.x;
    int co = (int)(i & 63);
    int m  = (int)(i >> 6);
    int ow = m & 31;
    int oh = (m >> 5) & 31;
    int n  = m >> 10;
    float acc = 0.f;
    #pragma unroll
    for (int c = 0; c < 3; ++c) {
        float mean = xstats[c];
        float rstd = rsqrtf(xstats[3 + c] + BN_EPS);
        #pragma unroll
        for (int kh = 0; kh < 3; ++kh) {
            int ih = oh - 1 + kh;
            if (ih < 0 || ih > 31) continue;
            #pragma unroll
            for (int kw = 0; kw < 3; ++kw) {
                int iw = ow - 1 + kw;
                if (iw < 0 || iw > 31) continue;
                float xv = (x[(((long)n * 3 + c) * 32 + ih) * 32 + iw] - mean) * rstd;
                acc += xv * w[(((long)co * 3 + c) * 3 + kh) * 3 + kw];
            }
        }
    }
    out[i] = acc;
}

__global__ __launch_bounds__(TPB) void k_bn_apply(
    const float* __restrict__ in, float* __restrict__ out,
    const float* __restrict__ stats, const float* __restrict__ g,
    const float* __restrict__ b, int C, long total)
{
    long i = (long)blockIdx.x * TPB + threadIdx.x;
    if (i >= total) return;
    int c = (int)(i % C);
    float r = rsqrtf(stats[C + c] + BN_EPS);
    out[i] = (in[i] - stats[c]) * r * g[c] + b[c];
}

__global__ __launch_bounds__(TPB) void k_bn_add_res(
    float* __restrict__ buf0, const float* __restrict__ in,
    const float* __restrict__ stats, const float* __restrict__ g,
    const float* __restrict__ b, int C, long total)
{
    long i = (long)blockIdx.x * TPB + threadIdx.x;
    if (i >= total) return;
    int c = (int)(i % C);
    float r = rsqrtf(stats[C + c] + BN_EPS);
    buf0[i] = buf0[i] + (in[i] - stats[c]) * r * g[c] + b[c];
}

__global__ __launch_bounds__(TPB) void k_bn_add2(
    float* __restrict__ buf0,
    const float* __restrict__ in1, const float* __restrict__ stA,
    const float* __restrict__ g1, const float* __restrict__ b1,
    const float* __restrict__ in2, const float* __restrict__ stB,
    const float* __restrict__ g2, const float* __restrict__ b2,
    int C, long total)
{
    long i = (long)blockIdx.x * TPB + threadIdx.x;
    if (i >= total) return;
    int c = (int)(i % C);
    float r1 = rsqrtf(stA[C + c] + BN_EPS);
    float r2 = rsqrtf(stB[C + c] + BN_EPS);
    buf0[i] = (in1[i] - stA[c]) * r1 * g1[c] + b1[c]
            + (in2[i] - stB[c]) * r2 * g2[c] + b2[c];
}

__global__ __launch_bounds__(TPB) void k_binarize_pad(
    const float* __restrict__ in, int8_t* __restrict__ out8,
    const float* __restrict__ s1, const float* __restrict__ s4,
    int C, int H, int W, long totalPad)
{
    long i = (long)blockIdx.x * TPB + threadIdx.x;
    if (i >= totalPad) return;
    int Wp = W + 2, Hp = H + 2;
    int c  = (int)(i % C);
    long r = i / C;
    int pw = (int)(r % Wp);
    long r2 = r / Wp;
    int ph = (int)(r2 % Hp);
    int n  = (int)(r2 / Hp);
    int8_t v = 0;
    if (ph >= 1 && ph <= H && pw >= 1 && pw <= W) {
        long src = (((long)n * H + (ph - 1)) * W + (pw - 1)) * C + c;
        float t = (s1[c] + 1.0f) * in[src] + s4[c] + 1e-6f;
        v = (t >= 0.f) ? (int8_t)1 : (int8_t)-1;
    }
    out8[i] = v;
}

__global__ __launch_bounds__(TPB) void k_wpack(
    const float* __restrict__ w, int8_t* __restrict__ wp,
    int Cin, int KHW, int Ksteps, int total)
{
    int i = blockIdx.x * TPB + threadIdx.x;
    if (i >= total) return;
    int tile = i >> 10;
    int r    = i & 1023;
    int lane = r >> 5;
    int v    = (r >> 2) & 7;
    int byt  = r & 3;
    int kt = tile % Ksteps;
    int nt = tile / Ksteps;
    int K  = kt * 64 + ((v >> 1) << 4) + ((lane >> 4) << 3) + ((v & 1) << 2) + byt;
    int N  = nt * 16 + (lane & 15);
    int c   = K % Cin;
    int tap = K / Cin;
    int kh = tap / KHW;
    int kw = tap - kh * KHW;
    float wv = w[(((long)N * Cin + c) * KHW + kh) * KHW + kw];
    wp[i] = (wv + 1e-6f >= 0.f) ? (int8_t)1 : (int8_t)-1;
}

template <int KHW, int STRIDE, int PAD>
__global__ __launch_bounds__(TPB) void k_gemm_bin(
    const int8_t* __restrict__ act8,
    const int8_t* __restrict__ wp,
    float* __restrict__ out,
    int Hp, int Wp, int C,
    int OH, int OW, int Cout,
    int NtPairs, int work)
{
    int wt = blockIdx.x * 8 + (threadIdx.x >> 5);
    if (wt >= work) return;
    int lane = threadIdx.x & 31;
    int hi   = lane >> 4;
    int ln   = lane & 15;
    int mt   = wt / NtPairs;
    int nt0  = (wt - mt * NtPairs) * 2;
    int Cb   = C >> 6;
    int Ksteps = KHW * KHW * Cb;

    int r0 = mt * 32 + ln;
    int r1 = r0 + 16;
    int ow0 = r0 % OW; int t0 = r0 / OW; int oh0 = t0 % OH; int n0 = t0 / OH;
    int ow1 = r1 % OW; int t1 = r1 / OW; int oh1 = t1 % OH; int n1 = t1 / OH;

    v8i acc00 = {0,0,0,0,0,0,0,0};
    v8i acc01 = acc00, acc10 = acc00, acc11 = acc00;

    const int8_t* bp0 = wp + ((long)nt0 * Ksteps) * 1024 + lane * 32;
    const int8_t* bp1 = bp0 + (long)Ksteps * 1024;

    #pragma unroll
    for (int kh = 0; kh < KHW; ++kh) {
        #pragma unroll
        for (int kw = 0; kw < KHW; ++kw) {
            const int8_t* p0 = act8
                + ((long)(n0 * Hp + oh0 * STRIDE + kh + 1 - PAD) * Wp
                   + (ow0 * STRIDE + kw + 1 - PAD)) * C + hi * 8;
            const int8_t* p1 = act8
                + ((long)(n1 * Hp + oh1 * STRIDE + kh + 1 - PAD) * Wp
                   + (ow1 * STRIDE + kw + 1 - PAD)) * C + hi * 8;
            for (int cb = 0; cb < Cb; ++cb) {
                v8i a0, a1;
                {
                    v2i q0 = *(const v2i*)(p0 +  0);
                    v2i q1 = *(const v2i*)(p0 + 16);
                    v2i q2 = *(const v2i*)(p0 + 32);
                    v2i q3 = *(const v2i*)(p0 + 48);
                    a0[0] = q0[0]; a0[1] = q0[1]; a0[2] = q1[0]; a0[3] = q1[1];
                    a0[4] = q2[0]; a0[5] = q2[1]; a0[6] = q3[0]; a0[7] = q3[1];
                }
                {
                    v2i q0 = *(const v2i*)(p1 +  0);
                    v2i q1 = *(const v2i*)(p1 + 16);
                    v2i q2 = *(const v2i*)(p1 + 32);
                    v2i q3 = *(const v2i*)(p1 + 48);
                    a1[0] = q0[0]; a1[1] = q0[1]; a1[2] = q1[0]; a1[3] = q1[1];
                    a1[4] = q2[0]; a1[5] = q2[1]; a1[6] = q3[0]; a1[7] = q3[1];
                }
                v8i b0 = *(const v8i*)bp0;
                v8i b1 = *(const v8i*)bp1;
                acc00 = __builtin_amdgcn_wmma_i32_16x16x64_iu8(true, a0, true, b0, acc00, false, false);
                acc01 = __builtin_amdgcn_wmma_i32_16x16x64_iu8(true, a0, true, b1, acc01, false, false);
                acc10 = __builtin_amdgcn_wmma_i32_16x16x64_iu8(true, a1, true, b0, acc10, false, false);
                acc11 = __builtin_amdgcn_wmma_i32_16x16x64_iu8(true, a1, true, b1, acc11, false, false);
                p0 += 64; p1 += 64; bp0 += 1024; bp1 += 1024;
            }
        }
    }

    int col0 = nt0 * 16 + ln;
    #pragma unroll
    for (int r = 0; r < 8; ++r) {
        long m0 = (long)mt * 32 + hi * 8 + r;
        long m1 = m0 + 16;
        out[m0 * Cout + col0]      = (float)acc00[r];
        out[m0 * Cout + col0 + 16] = (float)acc01[r];
        out[m1 * Cout + col0]      = (float)acc10[r];
        out[m1 * Cout + col0 + 16] = (float)acc11[r];
    }
}

__global__ __launch_bounds__(TPB) void k_relu_pool(
    const float* __restrict__ buf, float* __restrict__ pooled)
{
    int i = blockIdx.x * TPB + threadIdx.x;
    int c = i & 511;
    int n = i >> 9;
    float s = 0.f;
    #pragma unroll
    for (int j = 0; j < 16; ++j) {
        float v = buf[((long)n * 16 + j) * 512 + c];
        s += fmaxf(v, 0.f);
    }
    pooled[i] = s * (1.f / 16.f);
}

__global__ __launch_bounds__(TPB) void k_fc(
    const float* __restrict__ pooled, const float* __restrict__ fw,
    const float* __restrict__ fb, float* __restrict__ out)
{
    int i = blockIdx.x * TPB + threadIdx.x;
    if (i >= 2560) return;
    int j = i % 10;
    int n = i / 10;
    float s = fb[j];
    for (int k = 0; k < 512; ++k)
        s += pooled[n * 512 + k] * fw[j * 512 + k];
    out[i] = s;
}

extern "C" void kernel_launch(void* const* d_in, const int* in_sizes, int n_in,
                              void* d_out, int out_size, void* d_ws, size_t ws_size,
                              hipStream_t stream)
{
    (void)out_size; (void)ws_size;

    struct LB  { const float *s1, *s4; };
    struct BNP { const float *g, *b; };
    struct Blk { BNP bn1, bn2, bns; LB lb1, lb2, lbs; const float *w1, *w2, *ws; };

    Blk blk[8] = {};
    const float *xin = nullptr, *conv1w = nullptr, *topg = nullptr, *topb = nullptr;
    const float *fcw = nullptr, *fcb = nullptr;

    const int  cins[8]  = {64, 64, 64, 128, 128, 256, 256, 512};
    const int  couts[8] = {64, 64, 128, 128, 256, 256, 512, 512};
    const int  Hs[8]    = {32, 32, 32, 16, 16, 8, 8, 4};
    const bool dwn[8]   = {false, false, true, false, true, false, true, false};

    auto F = [&](int i) { return (const float*)d_in[i]; };
    bool insertion = (n_in > 0 && in_sizes[0] == 786432);
    int id = 0;
    if (!insertion) {
        for (int b = 0; b < 8; ++b) {
            blk[b].bn1.g = F(id++); blk[b].bn1.b = F(id++);
            blk[b].bn2.g = F(id++); blk[b].bn2.b = F(id++);
            if (dwn[b]) { blk[b].bns.g = F(id++); blk[b].bns.b = F(id++); }
            blk[b].lb1.s1 = F(id++); id += 2; blk[b].lb1.s4 = F(id++); id++;
            blk[b].lb2.s1 = F(id++); id += 2; blk[b].lb2.s4 = F(id++); id++;
            if (dwn[b]) { blk[b].lbs.s1 = F(id++); id += 2; blk[b].lbs.s4 = F(id++); id++; }
            blk[b].w1 = F(id++); blk[b].w2 = F(id++);
            if (dwn[b]) blk[b].ws = F(id++);
        }
        topg = F(id++); topb = F(id++);
        conv1w = F(id++); fcb = F(id++); fcw = F(id++); xin = F(id++);
    } else {
        xin = F(id++); conv1w = F(id++); topg = F(id++); topb = F(id++);
        for (int b = 0; b < 8; ++b) {
            blk[b].lb1.s1 = F(id++); id += 2; blk[b].lb1.s4 = F(id++); id++;
            blk[b].w1 = F(id++);
            blk[b].bn1.g = F(id++); blk[b].bn1.b = F(id++);
            blk[b].lb2.s1 = F(id++); id += 2; blk[b].lb2.s4 = F(id++); id++;
            blk[b].w2 = F(id++);
            blk[b].bn2.g = F(id++); blk[b].bn2.b = F(id++);
            if (dwn[b]) {
                blk[b].lbs.s1 = F(id++); id += 2; blk[b].lbs.s4 = F(id++); id++;
                blk[b].ws = F(id++);
                blk[b].bns.g = F(id++); blk[b].bns.b = F(id++);
            }
        }
        fcw = F(id++); fcb = F(id++);
    }

    char* W = (char*)d_ws;
    float*  buf0   = (float*)(W);
    float*  buf1   = (float*)(W + 0x04000000ull);
    float*  buf3   = (float*)(W + 0x08000000ull);
    int8_t* sgn    = (int8_t*)(W + 0x0A000000ull);
    int8_t* wp     = (int8_t*)(W + 0x0C000000ull);
    float*  statsA = (float*)(W + 0x0C400000ull);
    float*  statsB = statsA + 1024;
    float*  xstats = statsB + 1024;
    float*  pooled = (float*)(W + 0x0C500000ull);

    auto gblk = [](long total) { return (unsigned)((total + TPB - 1) / TPB); };

    auto binz = [&](const float* in, const LB& lb, int C, int H) {
        long totP = 256L * (H + 2) * (H + 2) * C;
        k_binarize_pad<<<gblk(totP), TPB, 0, stream>>>(in, sgn, lb.s1, lb.s4, C, H, H, totP);
    };

    auto bconv = [&](const float* wsrc, float* outp,
                     int Hin, int Cin, int OHW, int Cout, int khw, int strd) {
        int Ksteps = (Cin * khw * khw) >> 6;
        int wtot   = (Cout / 16) * Ksteps * 1024;
        k_wpack<<<gblk(wtot), TPB, 0, stream>>>(wsrc, wp, Cin, khw, Ksteps, wtot);
        int M = 256 * OHW * OHW;
        int NtPairs = Cout / 32;
        int work = (M / 32) * NtPairs;
        dim3 g((unsigned)((work + 7) / 8));
        int Hp = Hin + 2, Wp = Hin + 2;
        if (khw == 3 && strd == 1)
            k_gemm_bin<3, 1, 1><<<g, TPB, 0, stream>>>(sgn, wp, outp, Hp, Wp, Cin,
                                                       OHW, OHW, Cout, NtPairs, work);
        else if (khw == 3 && strd == 2)
            k_gemm_bin<3, 2, 1><<<g, TPB, 0, stream>>>(sgn, wp, outp, Hp, Wp, Cin,
                                                       OHW, OHW, Cout, NtPairs, work);
        else
            k_gemm_bin<1, 2, 0><<<g, TPB, 0, stream>>>(sgn, wp, outp, Hp, Wp, Cin,
                                                       OHW, OHW, Cout, NtPairs, work);
    };

    k_stats_nchw<<<3, TPB, 0, stream>>>(xin, 256, 3, 1024, xstats);
    k_conv1<<<gblk(16777216), TPB, 0, stream>>>(xin, conv1w, xstats, buf1);
    k_stats<<<64, TPB, 0, stream>>>(buf1, 262144, 64, statsA);
    k_bn_apply<<<gblk(16777216), TPB, 0, stream>>>(buf1, buf0, statsA, topg, topb, 64, 16777216);

    for (int b = 0; b < 8; ++b) {
        int Cin = cins[b], Cout = couts[b], H = Hs[b];
        if (!dwn[b]) {
            int M = 256 * H * H;
            long tot = (long)M * Cout;
            binz(buf0, blk[b].lb1, Cin, H);
            bconv(blk[b].w1, buf1, H, Cin, H, Cout, 3, 1);
            k_stats<<<Cout, TPB, 0, stream>>>(buf1, M, Cout, statsA);
            k_bn_apply<<<gblk(tot), TPB, 0, stream>>>(buf1, buf1, statsA, blk[b].bn1.g, blk[b].bn1.b, Cout, tot);
            binz(buf1, blk[b].lb2, Cout, H);
            bconv(blk[b].w2, buf1, H, Cout, H, Cout, 3, 1);
            k_stats<<<Cout, TPB, 0, stream>>>(buf1, M, Cout, statsA);
            k_bn_add_res<<<gblk(tot), TPB, 0, stream>>>(buf0, buf1, statsA, blk[b].bn2.g, blk[b].bn2.b, Cout, tot);
        } else {
            int H2 = H / 2;
            int Mo = 256 * H2 * H2;
            long toto = (long)Mo * Cout;
            binz(buf0, blk[b].lb1, Cin, H);
            bconv(blk[b].w1, buf1, H, Cin, H2, Cout, 3, 2);
            k_stats<<<Cout, TPB, 0, stream>>>(buf1, Mo, Cout, statsA);
            k_bn_apply<<<gblk(toto), TPB, 0, stream>>>(buf1, buf1, statsA, blk[b].bn1.g, blk[b].bn1.b, Cout, toto);
            binz(buf1, blk[b].lb2, Cout, H2);
            bconv(blk[b].w2, buf1, H2, Cout, H2, Cout, 3, 1);
            k_stats<<<Cout, TPB, 0, stream>>>(buf1, Mo, Cout, statsA);
            binz(buf0, blk[b].lbs, Cin, H);
            bconv(blk[b].ws, buf3, H, Cin, H2, Cout, 1, 2);
            k_stats<<<Cout, TPB, 0, stream>>>(buf3, Mo, Cout, statsB);
            k_bn_add2<<<gblk(toto), TPB, 0, stream>>>(buf0,
                buf1, statsA, blk[b].bn2.g, blk[b].bn2.b,
                buf3, statsB, blk[b].bns.g, blk[b].bns.b, Cout, toto);
        }
    }

    k_relu_pool<<<gblk(131072), TPB, 0, stream>>>(buf0, pooled);
    k_fc<<<gblk(2560), TPB, 0, stream>>>(pooled, fcw, fcb, (float*)d_out);
}
